// GCN_6193342840985
// MI455X (gfx1250) — matrix operand order repaired and checked by reading, not compiled, not run on hardware
//
#include <hip/hip_runtime.h>
#include <math.h>

// ---------------------------------------------------------------------------
// GCN chain per b: h=relu((A@X)@W1^T); h=relu((A@h)@W2^T); h=relu((A@h)@W3^T)
// out = l2norm(flatten(h) @ Wout^T + bout);  B=65536, N=D=20, Wout:[512,400]
//
// MI455X plan:
//  - big [B,400]x[400,512] projection on bf16 WMMA (f32 accum); Wout pre-
//    converted to bf16 [512][416] in workspace (L2-resident, 0.43MB << 192MB)
//  - tiny 20x20 chain in fp32 VALU, register-blocked rows + vector LDS loads
//    (~1 ds_load_b128 per 4 FMAs instead of 2 ds_load_b32 per FMA)
//  - A/X tiles (2 x 25.6KB contiguous) staged into LDS by the Tensor Data
//    Mover (tensor_load_to_lds + s_wait_tensorcnt), one descriptor per wave
// ---------------------------------------------------------------------------

typedef __attribute__((ext_vector_type(16))) __bf16 v16bf;
typedef __attribute__((ext_vector_type(8)))  float  v8f;
typedef __attribute__((ext_vector_type(4)))  unsigned int v4u;
typedef __attribute__((ext_vector_type(8)))  int    v8i;
typedef __attribute__((ext_vector_type(4)))  int    v4i;

#define MT      16          // batch elements per block
#define KP      416         // padded K (13 * 32)
#define KSTEPS  13
#define NOUT    512
#define NT_PER_WAVE 4       // 8 waves * 4 = 32 n-tiles = 512 cols

// ---- LDS layout (single carved buffer, offset 0 in group segment) ----
// [0, 13312)     : sHb : bf16 [16][416]  WMMA A-operand
// [13312, 90112) : sA(25600) sP(25600) sQ(25600); phase-2 overlays sOut
//                  fp32 [16][520] = 33280 here
// [90112, 93312) : sW  : W1(400f) W2(200f) W3(200f)
#define SM_HB   0
#define SM_R1   13312
#define SM_A    (SM_R1 + 0)
#define SM_P    (SM_R1 + 25600)
#define SM_Q    (SM_R1 + 51200)
#define SM_OUT  SM_R1
#define SM_W    90112
#define SM_TOT  93312
#define OUT_STRIDE 520

// -------------------- Wout fp32 -> bf16 [512][416] (zero pad) --------------
__global__ __launch_bounds__(256) void gcn_convert_wout(
    const float* __restrict__ Wout, unsigned short* __restrict__ WoutB) {
  int idx = blockIdx.x * 256 + threadIdx.x;       // 512*416 = 832*256
  int n = idx / KP;
  int k = idx - n * KP;
  __bf16 v = (k < 400) ? (__bf16)Wout[n * 400 + k] : (__bf16)0.0f;
  reinterpret_cast<__bf16*>(WoutB)[idx] = v;
}

// -------------------- TDM descriptor helpers (1-D copy, 4B elements) -------
// D# group0: count=1, lds_addr[63:32], global_addr[120:64], type=2 [127:126]
__device__ __forceinline__ v4u tdm_g0(unsigned lds_addr, unsigned long long ga) {
  v4u g;
  g[0] = 1u;                                    // count=1, user mode, no gather
  g[1] = lds_addr;
  g[2] = (unsigned)(ga & 0xFFFFFFFFu);
  g[3] = (unsigned)((ga >> 32) & 0x01FFFFFFu) | (2u << 30);
  return g;
}
// D# group1: data_size=4B, tensor_dim0=tile_dim0=n, tensor_dim1=tile_dim1=1
__device__ __forceinline__ v8i tdm_g1(unsigned n) {
  v8i g;
  g[0] = 0x00020000;                            // wg_mask=0, data_size=2 (4B)
  g[1] = (int)((n & 0xFFFFu) << 16);            // tensor_dim0[15:0] @ [63:48]
  g[2] = (int)((n >> 16) | (1u << 16));         // dim0[31:16]; tensor_dim1=1
  g[3] = (int)(n << 16);                        // tile_dim0 @ [127:112]
  g[4] = 1;                                     // tile_dim1=1; tile_dim2=0
  g[5] = (int)n;                                // tensor_dim0_stride[31:0]
  g[6] = 0;                                     // stride[47:32]=0; dim1_stride=0
  g[7] = 0;
  return g;
}
__device__ __forceinline__ void tdm_load_1d(unsigned lds_addr, const void* gp,
                                            unsigned n_dwords) {
  v4u g0 = tdm_g0(lds_addr, (unsigned long long)(uintptr_t)gp);
  v8i g1 = tdm_g1(n_dwords);
  v4i z4 = {0, 0, 0, 0};
#if __has_include(<hip/amd_detail/amd_gfx1250_TDM.h>)
  v8i z8 = {0, 0, 0, 0, 0, 0, 0, 0};
  __builtin_amdgcn_tensor_load_to_lds(g0, g1, z4, z4, z8, 0);
#else
  __builtin_amdgcn_tensor_load_to_lds(g0, g1, z4, z4, 0);
#endif
}

// -------------------- phase-1 register-blocked helpers ---------------------
template <int C>
__device__ __forceinline__ void ldrow(const float* p, float* r) {
  if constexpr (C % 4 == 0) {
#pragma unroll
    for (int v = 0; v < C / 4; ++v) {
      float4 t = reinterpret_cast<const float4*>(p)[v];
      r[4 * v] = t.x; r[4 * v + 1] = t.y; r[4 * v + 2] = t.z; r[4 * v + 3] = t.w;
    }
  } else {
#pragma unroll
    for (int v = 0; v < C / 2; ++v) {
      float2 t = reinterpret_cast<const float2*>(p)[v];
      r[2 * v] = t.x; r[2 * v + 1] = t.y;
    }
  }
}
template <int C>
__device__ __forceinline__ void strow(float* p, const float* r) {
  if constexpr (C % 4 == 0) {
#pragma unroll
    for (int v = 0; v < C / 4; ++v)
      reinterpret_cast<float4*>(p)[v] =
          make_float4(r[4 * v], r[4 * v + 1], r[4 * v + 2], r[4 * v + 3]);
  } else {
#pragma unroll
    for (int v = 0; v < C / 2; ++v)
      reinterpret_cast<float2*>(p)[v] = make_float2(r[2 * v], r[2 * v + 1]);
  }
}

// dst[e][i][0..C) = sum_o A[e][i][o] * src[e][o][0..C)   (row per task)
template <int C>
__device__ __forceinline__ void graphmm2(const float* sA, const float* sSrc,
                                         float* sDst, int tid) {
  for (int task = tid; task < MT * 20; task += 256) {
    int e = task / 20, i = task - e * 20;
    const float* Se = sSrc + e * 400;
    float arow[20];
    ldrow<20>(sA + e * 400 + i * 20, arow);
    float acc[C];
#pragma unroll
    for (int j = 0; j < C; ++j) acc[j] = 0.f;
#pragma unroll
    for (int o = 0; o < 20; ++o) {
      float srow[C];
      ldrow<C>(Se + o * C, srow);
      float a = arow[o];
#pragma unroll
      for (int j = 0; j < C; ++j) acc[j] = fmaf(a, srow[j], acc[j]);
    }
    strow<C>(sDst + e * 400 + i * C, acc);
  }
}

// dst[e][i][k] = relu(sum_d src[e][i][d] * W[k][d])   (row per task)
template <int C, int K>
__device__ __forceinline__ void linrelu2(const float* sSrc, const float* W,
                                         float* sDst, int tid) {
  for (int task = tid; task < MT * 20; task += 256) {
    int e = task / 20, i = task - e * 20;
    float srow[C];
    ldrow<C>(sSrc + e * 400 + i * C, srow);
    float acc[K];
#pragma unroll
    for (int k = 0; k < K; ++k) {
      float wrow[C];
      ldrow<C>(W + k * C, wrow);
      float s = 0.f;
#pragma unroll
      for (int d = 0; d < C; ++d) s = fmaf(srow[d], wrow[d], s);
      acc[k] = fmaxf(s, 0.f);
    }
    strow<K>(sDst + e * 400 + i * K, acc);
  }
}

// final layer: C=10, K=20, relu -> packed bf16 rows of sHb
__device__ __forceinline__ void lin3_bf16(const float* sSrc, const float* W3,
                                          __bf16* sHb, int tid) {
  for (int task = tid; task < MT * 20; task += 256) {
    int e = task / 20, i = task - e * 20;
    float srow[10];
    ldrow<10>(sSrc + e * 400 + i * 10, srow);
    unsigned* hrow = reinterpret_cast<unsigned*>(sHb + e * KP + i * 20);
#pragma unroll
    for (int k = 0; k < 20; k += 2) {
      float w0[10], w1[10];
      ldrow<10>(W3 + k * 10, w0);
      ldrow<10>(W3 + (k + 1) * 10, w1);
      float s0 = 0.f, s1 = 0.f;
#pragma unroll
      for (int d = 0; d < 10; ++d) {
        s0 = fmaf(srow[d], w0[d], s0);
        s1 = fmaf(srow[d], w1[d], s1);
      }
      unsigned short b0 = __builtin_bit_cast(unsigned short, (__bf16)fmaxf(s0, 0.f));
      unsigned short b1 = __builtin_bit_cast(unsigned short, (__bf16)fmaxf(s1, 0.f));
      hrow[k >> 1] = (unsigned)b0 | ((unsigned)b1 << 16);
    }
  }
}

// -------------------- fused main kernel ------------------------------------
__global__ __launch_bounds__(256) void gcn_main(
    const float* __restrict__ A, const float* __restrict__ X,
    const float* __restrict__ W1, const float* __restrict__ W2,
    const float* __restrict__ W3, const unsigned short* __restrict__ WoutBu,
    const float* __restrict__ bout, float* __restrict__ out) {
  __shared__ __align__(16) char smem[SM_TOT];
  float*  sA  = reinterpret_cast<float*>(smem + SM_A);
  float*  sP  = reinterpret_cast<float*>(smem + SM_P);
  float*  sQ  = reinterpret_cast<float*>(smem + SM_Q);
  float*  sW1 = reinterpret_cast<float*>(smem + SM_W);
  float*  sW2 = sW1 + 400;
  float*  sW3 = sW1 + 600;
  __bf16* sHb = reinterpret_cast<__bf16*>(smem + SM_HB);
  float*  sOut = reinterpret_cast<float*>(smem + SM_OUT);
  const __bf16* woutb = reinterpret_cast<const __bf16*>(WoutBu);

  const int tid  = threadIdx.x;
  const int wid  = tid >> 5;
  const int lane = tid & 31;

  // ---- stage A, X via Tensor Data Mover; weights via plain loads ----
  {
    // ISA 10.2: generic LDS pointer low 32 bits == wave LDS byte offset
    const unsigned ldsBase = (unsigned)(uintptr_t)(void*)smem;
    if (wid == 0) {
      tdm_load_1d(ldsBase + SM_A, A + (size_t)blockIdx.x * MT * 400, MT * 400);
    } else if (wid == 1) {
      tdm_load_1d(ldsBase + SM_P, X + (size_t)blockIdx.x * MT * 400, MT * 400);
    }
    for (int idx = tid; idx < 800; idx += 256) {
      if (idx < 400)      sW1[idx]       = W1[idx];
      else if (idx < 600) sW2[idx - 400] = W2[idx - 400];
      else                sW3[idx - 600] = W3[idx - 600];
    }
    if (wid < 2) __builtin_amdgcn_s_wait_tensorcnt(0);
  }
  __syncthreads();

  // ---- phase 1: fp32 chain, register-blocked rows ----
  graphmm2<20>(sA, sP, sQ, tid);        __syncthreads();  // t1 = A@X
  linrelu2<20, 20>(sQ, sW1, sP, tid);   __syncthreads();  // h1 [20,20]
  graphmm2<20>(sA, sP, sQ, tid);        __syncthreads();  // t2 = A@h1
  linrelu2<20, 10>(sQ, sW2, sP, tid);   __syncthreads();  // h2 [20,10]
  graphmm2<10>(sA, sP, sQ, tid);        __syncthreads();  // t3 = A@h2 [20,10]
  lin3_bf16(sQ, sW3, sHb, tid);                           // h3 -> bf16
  {
    int e = tid >> 4, r = tid & 15;                       // zero pad K=400..415
    sHb[e * KP + 400 + r] = (__bf16)0.0f;
  }
  __syncthreads();

  // ---- phase 2: [16 x 416] (bf16 LDS) x [416 x 512] (bf16 L2) WMMA ----
  const int l15 = lane & 15;
  const int halfsel = lane >> 4;

  union AF { v16bf v; uint4 q[2]; };
  v8f acc[NT_PER_WAVE] = {};
  const char* hbBytes = reinterpret_cast<const char*>(sHb);

  for (int k = 0; k < KSTEPS; ++k) {
    // A fragment, contiguous k order (same order as B):
    //  lanes 0-15: K = 32k+[0..15]; lanes 16-31: K = 32k+[16..31].
    AF a;
    int aoff = 2 * (l15 * KP + k * 32 + halfsel * 16);
    a.q[0] = *reinterpret_cast<const uint4*>(hbBytes + aoff);
    a.q[1] = *reinterpret_cast<const uint4*>(hbBytes + aoff + 16);

#pragma unroll
    for (int i = 0; i < NT_PER_WAVE; ++i) {
      // B fragment: lanes 0-15 hold K=32k+[0..15], lanes 16-31 K=32k+[16..31]
      int n = (wid * NT_PER_WAVE + i) * 16 + l15;
      const uint4* bp = reinterpret_cast<const uint4*>(
          woutb + (size_t)n * KP + k * 32 + halfsel * 16);
      AF b;
      b.q[0] = bp[0];
      b.q[1] = bp[1];
      acc[i] = __builtin_amdgcn_wmma_f32_16x16x32_bf16(
          false, a.v, false, b.v, (short)0, acc[i], false, false);
    }
  }

  // ---- stage D (+bout) to LDS: row m = halfsel*8 + r, col n = tile*16+l15 --
#pragma unroll
  for (int i = 0; i < NT_PER_WAVE; ++i) {
    int n = (wid * NT_PER_WAVE + i) * 16 + l15;
    float bo = bout[n];
#pragma unroll
    for (int r = 0; r < 8; ++r) {
      int m = halfsel * 8 + r;
      sOut[m * OUT_STRIDE + n] = acc[i][r] + bo;
    }
  }
  __syncthreads();

  // ---- l2 norm per row (2 rows per wave), wave32 shfl reduction ----
#pragma unroll
  for (int mi = 0; mi < 2; ++mi) {
    int m = wid * 2 + mi;
    float s = 0.f;
#pragma unroll
    for (int t = 0; t < NOUT / 32; ++t) {
      float v = sOut[m * OUT_STRIDE + lane + t * 32];
      s += v * v;
    }
#pragma unroll
    for (int off = 16; off >= 1; off >>= 1) s += __shfl_xor(s, off, 32);
    float rn = 1.0f / (sqrtf(s) + 1e-8f);
    float* orow = out + ((size_t)blockIdx.x * MT + m) * NOUT;
#pragma unroll
    for (int t = 0; t < NOUT / 32; ++t) {
      int n = lane + t * 32;
      orow[n] = sOut[m * OUT_STRIDE + n] * rn;
    }
  }
}

// ---------------------------------------------------------------------------
extern "C" void kernel_launch(void* const* d_in, const int* in_sizes, int n_in,
                              void* d_out, int out_size, void* d_ws, size_t ws_size,
                              hipStream_t stream) {
  const float* A    = (const float*)d_in[0];
  const float* X    = (const float*)d_in[1];
  const float* W1   = (const float*)d_in[2];
  const float* W2   = (const float*)d_in[3];
  const float* W3   = (const float*)d_in[4];
  const float* Wout = (const float*)d_in[5];
  const float* bout = (const float*)d_in[6];
  float* out = (float*)d_out;

  unsigned short* woutb = (unsigned short*)d_ws;   // 512*416*2 = 425984 bytes

  gcn_convert_wout<<<(NOUT * KP) / 256, 256, 0, stream>>>(Wout, woutb);

  const int B = 65536;
  gcn_main<<<B / MT, 256, 0, stream>>>(A, X, W1, W2, W3, woutb, bout, out);
}
